// MultiHeadAttention_87608742904540
// MI455X (gfx1250) — compile-verified, matrix operand order checked by reading, not run on hardware
//
#include <hip/hip_runtime.h>

// MI455X (gfx1250) MHA forward: f16 WMMA GEMMs + flash attention with
// async global->LDS staging (ASYNCcnt path) for K/V and weight tiles.

typedef __attribute__((ext_vector_type(16))) _Float16 v16h;
typedef __attribute__((ext_vector_type(8)))  _Float16 v8h;
typedef __attribute__((ext_vector_type(8)))  float    v8f;

#define D_MODEL 1024
#define N_HEAD  16
#define D_HEAD  64
#define BATCH   2
#define SEQ     2048
#define BH      (BATCH * N_HEAD)   // 32
#define ROWS    (BATCH * SEQ)      // 4096

union V16U { v16h v; v8h h[2]; };

// Load one 16x32 f16 A/B fragment from a row-major tile (LDS or global).
// ISA layout (CDNA5 ISA): lane holds row = lane%16; VGPR0-3: K=(lane/16)*8+{0..7},
// VGPR4-7: K=16+(lane/16)*8+{0..7}  -> two contiguous 16-byte loads per lane.
// B fragments use the same pattern when B^T is row-major ([N][K]).
// NOTE: stride (in f16 elems) must keep 16-byte row alignment (multiple of 8).
__device__ __forceinline__ v16h load_frag(const _Float16* base, int stride) {
  int lane = threadIdx.x & 31;
  int row  = lane & 15;
  int koff = (lane >> 4) << 3;
  const _Float16* p = base + (size_t)row * stride + koff;
  V16U u;
  u.h[0] = *(const v8h*)(p);
  u.h[1] = *(const v8h*)(p + 16);
  return u.v;
}

__device__ __forceinline__ v8f wmma_f16(v16h a, v16h b, v8f c) {
  return __builtin_amdgcn_wmma_f32_16x16x32_f16(
      /*neg_a=*/false, a, /*neg_b=*/false, b,
      /*c_mod=*/(short)0, c, /*reuse_a=*/false, /*reuse_b=*/false);
}

// Per-lane async 16B copy global -> LDS (tracked by ASYNCcnt, no VGPR data).
// Generic pointer to a __shared__ object truncates to the wave LDS offset
// (flat aperture mapping: low 32 bits are the LDS address).
__device__ __forceinline__ void async_copy16(void* lds_dst, const void* gsrc) {
  unsigned int loff = (unsigned int)(uintptr_t)lds_dst;
  unsigned long long ga = (unsigned long long)(uintptr_t)gsrc;
  asm volatile("global_load_async_to_lds_b128 %0, %1, off"
               :: "v"(loff), "v"(ga) : "memory");
}

__device__ __forceinline__ void async_wait0() {
  asm volatile("s_wait_asynccnt 0x0" ::: "memory");
}

// ---------------------------------------------------------------- weights f32->f16
__global__ __launch_bounds__(256) void convert_w(
    const float* __restrict__ Wq, const float* __restrict__ Wk,
    const float* __restrict__ Wv, _Float16* __restrict__ W16) {
  const size_t N = (size_t)D_MODEL * D_MODEL;
  size_t i = (size_t)blockIdx.x * blockDim.x + threadIdx.x;
  if (i < N)            W16[i] = (_Float16)Wq[i];
  else if (i < 2 * N)   W16[i] = (_Float16)Wk[i - N];
  else if (i < 3 * N)   W16[i] = (_Float16)Wv[i - 2 * N];
}

// ---------------------------------------------------------------- QKV projection
// Y = X @ W^T + b ; 128x128 tile per block, 8 waves each 32x64, K chunks of 32.
// A tile (f32->f16 convert) staged through VGPRs; B tile (already f16) staged
// with async global->LDS copies.
// mode 0 -> Qh[bh][s][64] (scaled 1/8); mode 1 -> Kh[bh][s][64];
// mode 2 -> Vt[bh][64][s] (transposed so P@V B-fragments are contiguous).
__global__ __launch_bounds__(256) void qkv_proj(
    const float* __restrict__ q, const float* __restrict__ k,
    const float* __restrict__ v, const _Float16* __restrict__ W16,
    const float* __restrict__ bq, const float* __restrict__ bk,
    const float* __restrict__ bv,
    _Float16* __restrict__ Qh, _Float16* __restrict__ Kh,
    _Float16* __restrict__ Vt) {
  __shared__ _Float16 As[128][48];   // 96B rows: 16B-aligned b128 chunks
  __shared__ _Float16 Bs[128][48];

  const int mode = blockIdx.z;
  const float* X = (mode == 0) ? q : (mode == 1) ? k : v;
  const float* bias = (mode == 0) ? bq : (mode == 1) ? bk : bv;
  const _Float16* W = W16 + (size_t)mode * D_MODEL * D_MODEL;

  const int tid = threadIdx.x;
  const int m0 = blockIdx.y * 128;         // token-row tile
  const int n0 = blockIdx.x * 128;         // output-col tile
  const int wid = tid >> 5;
  const int wr = wid & 3;                  // 4 row-blocks of 32
  const int wc = wid >> 2;                 // 2 col-blocks of 64
  const int lane = tid & 31;
  const int hf = lane >> 4;
  const int nl = lane & 15;

  v8f acc[2][4] = {};

  const int ldrow = tid >> 1;              // 0..127
  const int ldcol = (tid & 1) * 16;        // 0 or 16

  for (int k0 = 0; k0 < D_MODEL; k0 += 32) {
    // B tile: pure f16 copy -> async global->LDS (2 x 16B per thread = 8KB/block)
    {
      const _Float16* srcw = W + (size_t)(n0 + ldrow) * D_MODEL + k0 + ldcol;
      async_copy16(&Bs[ldrow][ldcol], srcw);
      async_copy16(&Bs[ldrow][ldcol + 8], srcw + 8);
    }
    // A tile: f32 -> f16 conversion through VGPRs
    {
      const float* src = X + (size_t)(m0 + ldrow) * D_MODEL + k0 + ldcol;
      v8h h0, h1;
#pragma unroll
      for (int e = 0; e < 8; ++e) h0[e] = (_Float16)src[e];
#pragma unroll
      for (int e = 0; e < 8; ++e) h1[e] = (_Float16)src[e + 8];
      *(v8h*)&As[ldrow][ldcol]     = h0;
      *(v8h*)&As[ldrow][ldcol + 8] = h1;
    }
    async_wait0();
    __syncthreads();

    v16h a[2], bf[4];
#pragma unroll
    for (int i = 0; i < 2; ++i) a[i] = load_frag(&As[wr * 32 + i * 16][0], 48);
#pragma unroll
    for (int j = 0; j < 4; ++j) bf[j] = load_frag(&Bs[wc * 64 + j * 16][0], 48);
#pragma unroll
    for (int i = 0; i < 2; ++i)
#pragma unroll
      for (int j = 0; j < 4; ++j) acc[i][j] = wmma_f16(a[i], bf[j], acc[i][j]);
    __syncthreads();
  }

  // epilogue: bias, scale, scatter to head-split f16 layouts
#pragma unroll
  for (int i = 0; i < 2; ++i)
#pragma unroll
    for (int j = 0; j < 4; ++j)
#pragma unroll
      for (int r = 0; r < 8; ++r) {
        int gm = m0 + wr * 32 + i * 16 + r + hf * 8;   // token row (0..4095)
        int gn = n0 + wc * 64 + j * 16 + nl;           // feature (0..1023)
        float val = acc[i][j][r] + bias[gn];
        int bb = gm >> 11, ss = gm & (SEQ - 1);
        int hh = gn >> 6,  dd = gn & (D_HEAD - 1);
        size_t bh = (size_t)bb * N_HEAD + hh;
        if (mode == 0)
          Qh[(bh * SEQ + ss) * D_HEAD + dd] = (_Float16)(val * 0.125f); // 1/sqrt(64)
        else if (mode == 1)
          Kh[(bh * SEQ + ss) * D_HEAD + dd] = (_Float16)val;
        else
          Vt[(bh * D_HEAD + dd) * SEQ + ss] = (_Float16)val;
      }
}

// ---------------------------------------------------------------- flash attention
// Block = 8 waves x 16 query rows = 128 rows. Each 64-key chunk of K and V is
// staged ONCE per block into LDS via async copies (8x less L2 traffic than
// per-wave global reads), then all waves consume it with ds_load_b128 fragment
// reads. scores = (Q K^T) * mask ; online softmax ; O += P V (P restaged in LDS).
__global__ __launch_bounds__(256) void attn(
    const _Float16* __restrict__ Qh, const _Float16* __restrict__ Kh,
    const _Float16* __restrict__ Vt, const float* __restrict__ mask,
    float* __restrict__ out) {
  __shared__ _Float16 Ks[64][72];      // 64 keys x 64 dh  (144B rows)
  __shared__ _Float16 Vs[64][72];      // 64 dh  x 64 keys (B^T layout)
  __shared__ _Float16 Pl[8][16][72];   // per-wave 16x64 P strip

  const int bh = blockIdx.y;                 // 0..31  (b*16 + h)
  const int bb = bh >> 4;
  const int tid = threadIdx.x;
  const int wid = tid >> 5;
  const int lane = tid & 31;
  const int hf = lane >> 4;
  const int nl = lane & 15;
  const int q0 = blockIdx.x * 128 + wid * 16;

  const _Float16* Qbase = Qh + ((size_t)bh * SEQ + q0) * D_HEAD;
  v16h qa[2];
  qa[0] = load_frag(Qbase, D_HEAD);        // dh 0..31
  qa[1] = load_frag(Qbase + 32, D_HEAD);   // dh 32..63

  float m[8], l[8];
  v8f o[4] = {};
#pragma unroll
  for (int r = 0; r < 8; ++r) { m[r] = -1e30f; l[r] = 0.0f; }

  const float* mrow = mask + (size_t)bb * SEQ * SEQ;   // mask[b][0][q][k]

  // staging map: each thread owns 32B of K and 32B of V per chunk
  const int srow = tid >> 2;              // 0..63
  const int scol = (tid & 3) * 16;        // 0/16/32/48

  for (int k0 = 0; k0 < SEQ; k0 += 64) {
    // ---- async stage K (rows = keys) and V (rows = dh, B^T) tiles
    {
      const _Float16* ksrc = Kh + ((size_t)bh * SEQ + k0 + srow) * D_HEAD + scol;
      async_copy16(&Ks[srow][scol], ksrc);
      async_copy16(&Ks[srow][scol + 8], ksrc + 8);
      const _Float16* vsrc = Vt + ((size_t)bh * D_HEAD + srow) * SEQ + k0 + scol;
      async_copy16(&Vs[srow][scol], vsrc);
      async_copy16(&Vs[srow][scol + 8], vsrc + 8);
    }
    // prefetch next mask chunk while tiles land
    if (k0 + 64 < SEQ)
      __builtin_prefetch(&mrow[(size_t)(q0 + hf * 8) * SEQ + k0 + 64 + nl * 4], 0, 1);
    async_wait0();
    __syncthreads();

    // ---- scores 16x64 as 4 C-fragments (Dh=64 -> 2 WMMAs each)
    v8f sc[4];
#pragma unroll
    for (int j = 0; j < 4; ++j) {
      v16h kb0 = load_frag(&Ks[j * 16][0], 72);
      v16h kb1 = load_frag(&Ks[j * 16][32], 72);
      v8f cz = {};
      cz = wmma_f16(qa[0], kb0, cz);
      cz = wmma_f16(qa[1], kb1, cz);
      sc[j] = cz;
    }

    // ---- multiplicative mask + per-row chunk max (row r+8*hf in this lane)
    float cm[8];
#pragma unroll
    for (int r = 0; r < 8; ++r) cm[r] = -1e30f;
#pragma unroll
    for (int j = 0; j < 4; ++j)
#pragma unroll
      for (int r = 0; r < 8; ++r) {
        int qrow = q0 + r + hf * 8;
        int kcol = k0 + j * 16 + nl;
        float sv = sc[j][r] * mrow[(size_t)qrow * SEQ + kcol];
        sc[j][r] = sv;
        cm[r] = fmaxf(cm[r], sv);
      }
#pragma unroll
    for (int r = 0; r < 8; ++r) {   // 16-lane max (stays within half)
      float x = cm[r];
      x = fmaxf(x, __shfl_xor(x, 1, 32));
      x = fmaxf(x, __shfl_xor(x, 2, 32));
      x = fmaxf(x, __shfl_xor(x, 4, 32));
      x = fmaxf(x, __shfl_xor(x, 8, 32));
      cm[r] = x;
    }

    // ---- online-softmax update
    float rs[8];
#pragma unroll
    for (int r = 0; r < 8; ++r) {
      float mn = fmaxf(m[r], cm[r]);
      float scale = __expf(m[r] - mn);
      m[r] = mn;
      l[r] *= scale;
#pragma unroll
      for (int j = 0; j < 4; ++j) o[j][r] *= scale;
      rs[r] = 0.0f;
    }
#pragma unroll
    for (int j = 0; j < 4; ++j)
#pragma unroll
      for (int r = 0; r < 8; ++r) {
        float p = __expf(sc[j][r] - m[r]);
        rs[r] += p;
        Pl[wid][r + hf * 8][j * 16 + nl] = (_Float16)p;   // C-layout -> row-major
      }
#pragma unroll
    for (int r = 0; r < 8; ++r) {   // 16-lane sum
      float x = rs[r];
      x += __shfl_xor(x, 1, 32);
      x += __shfl_xor(x, 2, 32);
      x += __shfl_xor(x, 4, 32);
      x += __shfl_xor(x, 8, 32);
      l[r] += x;
    }

    // ---- O += P(16x64) @ V(64x64) from LDS tiles
    v16h pa[2];
    pa[0] = load_frag(&Pl[wid][0][0], 72);    // keys k0+0..31
    pa[1] = load_frag(&Pl[wid][0][32], 72);   // keys k0+32..63
#pragma unroll
    for (int j = 0; j < 4; ++j) {
      v16h vb0 = load_frag(&Vs[j * 16][0], 72);
      v16h vb1 = load_frag(&Vs[j * 16][32], 72);
      o[j] = wmma_f16(pa[0], vb0, o[j]);
      o[j] = wmma_f16(pa[1], vb1, o[j]);
    }
    __syncthreads();   // keep tiles live until all waves consumed them
  }

  // ---- normalize + store (reference keeps raw [b,h,s,dh] flat order)
  float* ob = out + ((size_t)bh * SEQ + q0) * D_HEAD;
#pragma unroll
  for (int j = 0; j < 4; ++j)
#pragma unroll
    for (int r = 0; r < 8; ++r) {
      int row = r + hf * 8;
      ob[(size_t)row * D_HEAD + j * 16 + nl] = o[j][r] / l[r];
    }
}

// ---------------------------------------------------------------- launch
extern "C" void kernel_launch(void* const* d_in, const int* in_sizes, int n_in,
                              void* d_out, int out_size, void* d_ws, size_t ws_size,
                              hipStream_t stream) {
  const float* q    = (const float*)d_in[0];
  const float* k    = (const float*)d_in[1];
  const float* v    = (const float*)d_in[2];
  const float* mask = (const float*)d_in[3];
  const float* Wq   = (const float*)d_in[4];
  const float* bq   = (const float*)d_in[5];
  const float* Wk   = (const float*)d_in[6];
  const float* bk   = (const float*)d_in[7];
  const float* Wv   = (const float*)d_in[8];
  const float* bv   = (const float*)d_in[9];

  // workspace: 6MB W16 | 8MB Qh | 8MB Kh | 8MB Vt  (30MB total)
  char* ws = (char*)d_ws;
  _Float16* W16 = (_Float16*)(ws);
  _Float16* Qh  = (_Float16*)(ws + (size_t)6  * 1024 * 1024);
  _Float16* Kh  = (_Float16*)(ws + (size_t)14 * 1024 * 1024);
  _Float16* Vt  = (_Float16*)(ws + (size_t)22 * 1024 * 1024);

  convert_w<<<dim3((3u * D_MODEL * D_MODEL) / 256), dim3(256), 0, stream>>>(
      Wq, Wk, Wv, W16);
  qkv_proj<<<dim3(D_MODEL / 128, ROWS / 128, 3), dim3(256), 0, stream>>>(
      q, k, v, W16, bq, bk, bv, Qh, Kh, Vt);
  attn<<<dim3(SEQ / 128, BH), dim3(256), 0, stream>>>(
      Qh, Kh, Vt, mask, (float*)d_out);
}
